// IELin_14791867368254
// MI455X (gfx1250) — matrix operand order repaired and checked by reading, compiled, not run on hardware
//
#include <hip/hip_runtime.h>

// Block-diagonal linear: out[:, s_l:s_l+w_l] = x[:, s_l:s_l+w_l] @ W_l^T
// bf16x3 split-precision GEMM on v_wmma_f32_16x16x32_bf16 (CDNA5, wave32).
//
// Two kernels:
//  1) split_weights: one-time (per call) split of all W_l fp32 -> bf16 hi/lo
//     planes in workspace (~10.5 MB), removing redundant per-tile conversion.
//  2) GEMM: X split on the fly; W planes copied global->LDS with
//     GLOBAL_LOAD_ASYNC_TO_LDS_B128 (ASYNCcnt); double-buffered LDS,
//     one barrier per K-chunk, register prefetch of next X tile.

#define TOTAL_W 4096
#define N_ROWS  32768
#define BM 128
#define BN 128
#define BK 32
#define LDK 40   // LDS row pitch in ushorts: 80 bytes (16B aligned, 20-dword bank stride)
#define W_ELEMS 2752512  // sum of w_l^2 over the 8 blocks

typedef __attribute__((ext_vector_type(16))) __bf16 v16bf;
typedef __attribute__((ext_vector_type(8)))  float  v8f;
typedef __attribute__((ext_vector_type(4)))  int    v4i;

typedef __attribute__((address_space(1))) v4i* v4i_gptr;
typedef __attribute__((address_space(3))) v4i* v4i_sptr;

struct WPtrs { const float* W[8]; };

// 32 column tiles of 128 spanning the 8 blocks (1,1,3,3,5,5,7,7 tiles each)
__constant__ int c_tile_block[32] = {0,1, 2,2,2, 3,3,3, 4,4,4,4,4, 5,5,5,5,5,
                                     6,6,6,6,6,6,6, 7,7,7,7,7,7,7};
__constant__ int c_tile_ctile[32] = {0,0, 0,1,2, 0,1,2, 0,1,2,3,4, 0,1,2,3,4,
                                     0,1,2,3,4,5,6, 0,1,2,3,4,5,6};
__constant__ int c_starts[8] = {0,128,256,640,1024,1664,2304,3200};
__constant__ int c_widths[8] = {128,128,384,384,640,640,896,896};
// element offset of each block's weight plane inside whi/wlo
__constant__ unsigned c_weoff[8] = {0, 16384, 32768, 180224, 327680,
                                    737280, 1146880, 1949696};

union FragBF {
    v16bf v;
    uint4 q[2];
};

// Split fp32 into bf16 hi + bf16 lo (RNE); hi + lo ~ fp32.
__device__ __forceinline__ void split_bf16(float x, unsigned short& h, unsigned short& l) {
    unsigned int u  = __float_as_uint(x);
    unsigned int rh = u + 0x7FFFu + ((u >> 16) & 1u);
    h = (unsigned short)(rh >> 16);
    float hf = __uint_as_float(((unsigned int)h) << 16);
    float r  = x - hf;
    unsigned int ur = __float_as_uint(r);
    unsigned int rl = ur + 0x7FFFu + ((ur >> 16) & 1u);
    l = (unsigned short)(rl >> 16);
}

// 16-byte async global->LDS copy (ASYNCcnt-tracked on gfx1250).
__device__ __forceinline__ void async_copy_b128(const void* g, void* l) {
#if __has_builtin(__builtin_amdgcn_global_load_async_to_lds_b128)
    __builtin_amdgcn_global_load_async_to_lds_b128(
        (v4i_gptr)(void*)(g), (v4i_sptr)(l), 0, 0);
#else
    *(uint4*)l = *(const uint4*)g;   // VGPR-staged fallback
#endif
}

__device__ __forceinline__ void wait_async0() {
#if __has_builtin(__builtin_amdgcn_s_wait_asynccnt)
    __builtin_amdgcn_s_wait_asynccnt(0);
#else
    asm volatile("s_wait_asynccnt 0x0" ::: "memory");
#endif
}

// ---- Prep: split every W_l into bf16 hi/lo planes in workspace ----
__global__ __launch_bounds__(256) void split_weights_kernel(
        WPtrs wp, unsigned short* __restrict__ whi, unsigned short* __restrict__ wlo) {
    const int l = blockIdx.y;
    const int w = c_widths[l];
    const unsigned n = (unsigned)w * (unsigned)w;
    const unsigned idx = blockIdx.x * 256u + threadIdx.x;
    if (idx >= n) return;
    unsigned short h, lo;
    split_bf16(wp.W[l][idx], h, lo);
    const unsigned off = c_weoff[l];
    whi[off + idx] = h;
    wlo[off + idx] = lo;
}

// ---- Main GEMM ----
__global__ __launch_bounds__(256) void ielin_bf16x3_kernel(
        const float* __restrict__ x,
        const unsigned short* __restrict__ whi,
        const unsigned short* __restrict__ wlo,
        float* __restrict__ out) {
    // double-buffered planes: 8 * 10240 B = 80 KB of the 320 KB WGP pool
    __shared__ alignas(16) unsigned short sAh[2][BM * LDK];
    __shared__ alignas(16) unsigned short sAl[2][BM * LDK];
    __shared__ alignas(16) unsigned short sBh[2][BN * LDK];
    __shared__ alignas(16) unsigned short sBl[2][BN * LDK];

    const int t       = blockIdx.x;            // global column tile 0..31
    const int l       = c_tile_block[t];
    const int ct      = c_tile_ctile[t];
    const int kStart  = c_starts[l];
    const int KB      = c_widths[l];
    const int colLoc  = ct * BN;
    const int colGlob = kStart + colLoc;
    const int rowBase = blockIdx.y * BM;

    const int tid  = threadIdx.x;
    const int lane = tid & 31;
    const int wv   = tid >> 5;
    const int wm   = wv & 3;       // rows wm*32 .. +32
    const int wn   = wv >> 2;      // cols wn*64 .. +64
    const int half = lane >> 4;
    const int ln   = lane & 15;

    const float* xbase = x + (size_t)rowBase * TOTAL_W + kStart;
    const unsigned weoff = c_weoff[l];
    const unsigned short* bhin = whi + weoff + (size_t)colLoc * KB;
    const unsigned short* blin = wlo + weoff + (size_t)colLoc * KB;

    // per-thread staging coords
    const int arow = tid >> 3, akv = tid & 7;    // A: 8 threads per 32-float row
    const int brow0 = tid >> 2, bseg0 = tid & 3; // B: 4 x b128 segs per 32-bf16 row

    v8f acc[2][4] = {};
    float4 fxr[4];                // X prefetch registers (next chunk)
    const int nchunks = KB / BK;

    // ---- prologue: async B(0) into buf0, X(0) into regs ----
    #pragma unroll
    for (int i = 0; i < 2; ++i) {
        int row = brow0 + i * 64, seg = bseg0;
        int o = row * LDK + seg * 8;
        size_t g = (size_t)row * KB + seg * 8;
        async_copy_b128(bhin + g, &sBh[0][o]);
        async_copy_b128(blin + g, &sBl[0][o]);
    }
    #pragma unroll
    for (int i = 0; i < 4; ++i) {
        int row = arow + i * 32;
        fxr[i] = *(const float4*)(xbase + (size_t)row * TOTAL_W + akv * 4);
    }

    for (int c = 0; c < nchunks; ++c) {
        const int cur = c & 1;

        // ---- convert prefetched X regs -> bf16 hi/lo planes (buffer cur) ----
        #pragma unroll
        for (int i = 0; i < 4; ++i) {
            int row = arow + i * 32;
            ushort4 xh, xl;
            split_bf16(fxr[i].x, xh.x, xl.x); split_bf16(fxr[i].y, xh.y, xl.y);
            split_bf16(fxr[i].z, xh.z, xl.z); split_bf16(fxr[i].w, xh.w, xl.w);
            int o = row * LDK + akv * 4;
            *(ushort4*)&sAh[cur][o] = xh;
            *(ushort4*)&sAl[cur][o] = xl;
        }
        wait_async0();            // B(c) landed in LDS
        __syncthreads();          // one barrier per chunk (double-buffered)

        // ---- issue next chunk's B async copies + X register prefetch ----
        if (c + 1 < nchunks) {
            const int nxt = cur ^ 1;
            const int k1 = (c + 1) * BK;
            #pragma unroll
            for (int i = 0; i < 2; ++i) {
                int row = brow0 + i * 64, seg = bseg0;
                int o = row * LDK + seg * 8;
                size_t g = (size_t)row * KB + k1 + seg * 8;
                async_copy_b128(bhin + g, &sBh[nxt][o]);
                async_copy_b128(blin + g, &sBl[nxt][o]);
            }
            #pragma unroll
            for (int i = 0; i < 4; ++i) {
                int row = arow + i * 32;
                fxr[i] = *(const float4*)(xbase + (size_t)row * TOTAL_W + k1 + akv * 4);
            }
        }

        // ---- fragments from buffer cur ----
        const unsigned short* aH = sAh[cur];
        const unsigned short* aL = sAl[cur];
        const unsigned short* bH = sBh[cur];
        const unsigned short* bL = sBl[cur];

        FragBF ah[2], al[2], bh[4], bl[4];
        #pragma unroll
        for (int mt = 0; mt < 2; ++mt) {
            int r = (wm * 32 + mt * 16 + ln) * LDK + half * 8;
            const uint4* qa = (const uint4*)&aH[r];
            ah[mt].q[0] = qa[0]; ah[mt].q[1] = qa[2];
            const uint4* ql = (const uint4*)&aL[r];
            al[mt].q[0] = ql[0]; al[mt].q[1] = ql[2];
        }
        #pragma unroll
        for (int nt = 0; nt < 4; ++nt) {
            int r = (wn * 64 + nt * 16 + ln) * LDK + half * 8;
            const uint4* qb = (const uint4*)&bH[r];
            bh[nt].q[0] = qb[0]; bh[nt].q[1] = qb[2];
            const uint4* ql = (const uint4*)&bL[r];
            bl[nt].q[0] = ql[0]; bl[nt].q[1] = ql[2];
        }

        // ---- bf16x3: acc += Ah*Bh + Ah*Bl + Al*Bh ----
        #pragma unroll
        for (int mt = 0; mt < 2; ++mt)
            #pragma unroll
            for (int nt = 0; nt < 4; ++nt) {
                acc[mt][nt] = __builtin_amdgcn_wmma_f32_16x16x32_bf16(
                    false, ah[mt].v, false, bh[nt].v, (short)0, acc[mt][nt], false, false);
                acc[mt][nt] = __builtin_amdgcn_wmma_f32_16x16x32_bf16(
                    false, ah[mt].v, false, bl[nt].v, (short)0, acc[mt][nt], false, false);
                acc[mt][nt] = __builtin_amdgcn_wmma_f32_16x16x32_bf16(
                    false, al[mt].v, false, bh[nt].v, (short)0, acc[mt][nt], false, false);
            }
    }

    // ---- epilogue: D layout -> row = v + 8*half, col = lane%16 ----
    #pragma unroll
    for (int mt = 0; mt < 2; ++mt) {
        int gr0 = rowBase + wm * 32 + mt * 16 + half * 8;
        #pragma unroll
        for (int nt = 0; nt < 4; ++nt) {
            int gc = colGlob + wn * 64 + nt * 16 + ln;
            float* op = out + (size_t)gr0 * TOTAL_W + gc;
            #pragma unroll
            for (int v = 0; v < 8; ++v)
                op[(size_t)v * TOTAL_W] = acc[mt][nt][v];
        }
    }
}

extern "C" void kernel_launch(void* const* d_in, const int* in_sizes, int n_in,
                              void* d_out, int out_size, void* d_ws, size_t ws_size,
                              hipStream_t stream) {
    (void)in_sizes; (void)n_in; (void)out_size; (void)ws_size;
    const float* x = (const float*)d_in[0];
    WPtrs wp;
    for (int i = 0; i < 8; ++i) wp.W[i] = (const float*)d_in[1 + i];
    float* out = (float*)d_out;

    // workspace: bf16 hi plane then lo plane, W_ELEMS ushorts each (~10.5 MB)
    unsigned short* whi = (unsigned short*)d_ws;
    unsigned short* wlo = whi + W_ELEMS;

    // 1) split weights (largest block: 896*896 / 256 = 3136 blocks)
    split_weights_kernel<<<dim3(3136, 8), 256, 0, stream>>>(wp, whi, wlo);
    // 2) GEMM: 32 column tiles x 256 row tiles
    ielin_bf16x3_kernel<<<dim3(32, N_ROWS / BM), 256, 0, stream>>>(x, whi, wlo, out);
}
